// AngleSoftmax_62148176773303
// MI455X (gfx1250) — matrix operand order repaired and checked by reading, not compiled, not run on hardware
//
#include <hip/hip_runtime.h>
#include <hip/hip_bf16.h>
#include <stdint.h>

// ---------------------------------------------------------------------------
// A-Softmax (SphereFace, m=4) forward for MI455X / gfx1250.
//   B=512, D=256, C=100000, lambda=1000.
// GEMM done with v_wmma_f32_16x16x32_bf16 using an error-compensated bf16
// split (hi/lo) of both operands: 3 bf16 WMMAs reproduce the f32 product to
// ~2^-16 relative accuracy while running at bf16 matrix rate.
// ---------------------------------------------------------------------------

typedef __attribute__((ext_vector_type(8)))  float  v8f;
typedef __attribute__((ext_vector_type(16))) __bf16 v16bf;

#define Dk 256      // feature dim (K)
#define Cc 100000   // classes (N)
#define Bb 512      // batch (M)
#define TM 256      // workgroup M tile
#define TN 32       // workgroup N tile (2 N-waves x 16)

union BF16x16 {
    v16bf          v;
    unsigned short s[16];
    float4         f4[2];
};

__device__ __forceinline__ unsigned short bf_hi(float v) {
    return (unsigned short)(__float_as_uint(v) >> 16);   // truncation split
}
__device__ __forceinline__ float bf_hi_f(float v) {
    return __uint_as_float(__float_as_uint(v) & 0xffff0000u);
}

// ---------------------------------------------------------------------------
// Prep: inverse L2 norm of each weight row (one wave32 per row).
// Side effect: streams the whole 102 MB weight matrix through L2 (192 MB),
// so the GEMM kernel's weight reads mostly hit L2.
// ---------------------------------------------------------------------------
__global__ __launch_bounds__(256) void wnorm_k(const float* __restrict__ w,
                                               float* __restrict__ invn) {
    const int row  = blockIdx.x * 8 + (threadIdx.x >> 5);
    const int lane = threadIdx.x & 31;
    const float4* p = (const float4*)(w + (size_t)row * Dk);
    float4 a = p[lane];
    float4 b = p[lane + 32];
    float s = a.x*a.x + a.y*a.y + a.z*a.z + a.w*a.w
            + b.x*b.x + b.y*b.y + b.z*b.z + b.w*b.w;
#pragma unroll
    for (int off = 16; off > 0; off >>= 1) s += __shfl_xor(s, off, 32);
    if (lane == 0) invn[row] = 1.0f / sqrtf(s);
}

// Prep: ||x_b|| and 1/||x_b|| (one wave32 per row).
__global__ __launch_bounds__(256) void xnorm_k(const float* __restrict__ x,
                                               float* __restrict__ xl,
                                               float* __restrict__ ixl) {
    const int row  = blockIdx.x * 8 + (threadIdx.x >> 5);
    const int lane = threadIdx.x & 31;
    const float4* p = (const float4*)(x + (size_t)row * Dk);
    float4 a = p[lane];
    float4 b = p[lane + 32];
    float s = a.x*a.x + a.y*a.y + a.z*a.z + a.w*a.w
            + b.x*b.x + b.y*b.y + b.z*b.z + b.w*b.w;
#pragma unroll
    for (int off = 16; off > 0; off >>= 1) s += __shfl_xor(s, off, 32);
    if (lane == 0) {
        float l = sqrtf(s);
        xl[row]  = l;
        ixl[row] = 1.0f / l;
    }
}

// ---------------------------------------------------------------------------
// Main fused GEMM + margin epilogue.
// Grid: (Cc/TN = 3125, Bb/TM = 2). Block: 256 threads = 8 wave32.
// Wave (mwave 0..3, nwave 0..1) owns a 64x16 output tile (4 accum frags).
// LDS (dynamic, 265216 B): x slice pre-split to bf16 hi/lo in WMMA A layout
// (128 KB + 128 KB) + per-row xlen / 1/xlen / y (3 KB).
// ---------------------------------------------------------------------------
__global__ __launch_bounds__(256) void asoft_main(
        const float* __restrict__ x, const float* __restrict__ w,
        const int* __restrict__ y,
        const float* __restrict__ inv_wl,
        const float* __restrict__ xlen, const float* __restrict__ inv_xlen,
        float* __restrict__ out) {
    extern __shared__ char smem[];
    unsigned short* aHi = (unsigned short*)smem;   // 16 mfrag * 8 ks * 32 lane * 16 = 65536 bf16
    unsigned short* aLo = aHi + 65536;             // second 128 KB
    float* sXlen = (float*)(smem + 262144);
    float* sIxl  = sXlen + TM;
    int*   sY    = (int*)(sIxl + TM);

    const int tid = threadIdx.x;
    const int m0  = blockIdx.y * TM;
    const int c0  = blockIdx.x * TN;

    // ---- Phase 1: build the LDS A image (bf16 hi/lo, contiguous k = 16h + s order) ----
    {
        const int m     = tid;            // local row 0..255 (one row per thread)
        const int mfrag = m >> 4;
        const int mr    = m & 15;
        const float* xr = x + (size_t)(m0 + m) * Dk;
#pragma unroll 4
        for (int k = 0; k < Dk; k += 4) {
            float4 v4 = *(const float4*)(xr + k);
            const int kstep = k >> 5;
            const int kk    = k & 31;
            const int g     = kk >> 4;                       // lane group (K 0-15 vs 16-31)
            const int idx   = kk & 15;                       // element index within lane (4 consecutive)
            const int base  = ((mfrag * 8 + kstep) * 32 + (mr + (g << 4))) * 16 + idx;
            const float f0 = v4.x, f1 = v4.y, f2 = v4.z, f3 = v4.w;
            const unsigned short h0 = bf_hi(f0), h1 = bf_hi(f1), h2 = bf_hi(f2), h3 = bf_hi(f3);
            const unsigned short l0 = bf_hi(f0 - bf_hi_f(f0));
            const unsigned short l1 = bf_hi(f1 - bf_hi_f(f1));
            const unsigned short l2 = bf_hi(f2 - bf_hi_f(f2));
            const unsigned short l3 = bf_hi(f3 - bf_hi_f(f3));
            *(uint2*)(aHi + base) =
                make_uint2((unsigned)h0 | ((unsigned)h1 << 16), (unsigned)h2 | ((unsigned)h3 << 16));
            *(uint2*)(aLo + base) =
                make_uint2((unsigned)l0 | ((unsigned)l1 << 16), (unsigned)l2 | ((unsigned)l3 << 16));
        }
        sXlen[m] = xlen[m0 + m];
        sIxl[m]  = inv_xlen[m0 + m];
        sY[m]    = y[m0 + m];
    }
    __syncthreads();

    // ---- Phase 2: K loop with 3-product split-bf16 WMMA -----------------
    const int lane  = tid & 31;
    const int wave  = tid >> 5;
    const int mwave = wave & 3;   // 4 M-groups of 64 rows
    const int nwave = wave >> 2;  // 2 N-groups of 16 cols
    const int nloc  = lane & 15;  // column within N-frag (B/C/D lane layout)
    const int khalf = lane >> 4;  // K half (this code's contiguous B k order: lanes 0-15 K 0-15, 16-31 K 16-31)
    const int myc   = c0 + nwave * 16 + nloc;

    const float  invw = inv_wl[myc];
    const float* wrow = w + (size_t)myc * Dk + (khalf << 4);

    v8f acc[4] = {};
    for (int ks = 0; ks < 8; ++ks) {
        // B fragment: this lane's 16 consecutive weights (its column, its K-half)
        union { float4 f4[4]; float f[16]; } q;
        const float* wp = wrow + (ks << 5);
        q.f4[0] = *(const float4*)(wp + 0);
        q.f4[1] = *(const float4*)(wp + 4);
        q.f4[2] = *(const float4*)(wp + 8);
        q.f4[3] = *(const float4*)(wp + 12);
        if (ks < 7) __builtin_prefetch(wp + 32, 0, 1);   // global_prefetch_b8 next slab

        BF16x16 bhi, blo;
#pragma unroll
        for (int i = 0; i < 16; ++i) {
            const float v = q.f[i] * invw;               // row-normalized weight
            bhi.s[i] = bf_hi(v);
            blo.s[i] = bf_hi(v - bf_hi_f(v));
        }

#pragma unroll
        for (int f = 0; f < 4; ++f) {
            const int   mfrag = (mwave << 2) + f;
            const char* ab    = smem + (size_t)(((mfrag * 8 + ks) * 32 + lane) << 5);
            BF16x16 ah, al;
            ah.f4[0] = *(const float4*)(ab);
            ah.f4[1] = *(const float4*)(ab + 16);
            al.f4[0] = *(const float4*)(ab + 131072);
            al.f4[1] = *(const float4*)(ab + 131072 + 16);
            acc[f] = __builtin_amdgcn_wmma_f32_16x16x32_bf16(
                         false, ah.v, false, bhi.v, (short)0, acc[f], false, false);
            acc[f] = __builtin_amdgcn_wmma_f32_16x16x32_bf16(
                         false, ah.v, false, blo.v, (short)0, acc[f], false, false);
            acc[f] = __builtin_amdgcn_wmma_f32_16x16x32_bf16(
                         false, al.v, false, bhi.v, (short)0, acc[f], false, false);
        }
    }

    // ---- Phase 3: A-Softmax margin epilogue + NT stores ------------------
    // k = floor(4*acos(c)/pi) via threshold counting (no acos needed):
    // boundaries at cos(pi/4)=r2, cos(pi/2)=0, cos(3pi/4)=-r2, cos(pi)=-1.
    const float r2 = 0.70710678118654752f;
#pragma unroll
    for (int f = 0; f < 4; ++f) {
        const int rbase = ((mwave << 2) + f) * 16 + (khalf << 3);  // local row of acc[f][0]
#pragma unroll
        for (int j = 0; j < 8; ++j) {
            const int   rl  = rbase + j;
            const float xl  = sXlen[rl];
            const float ixl = sIxl[rl];
            const int   yy  = sY[rl];
            float co = acc[f][j] * ixl;
            co = fminf(fmaxf(co, -1.0f), 1.0f);
            const float c2   = co * co;
            const float cosm = 8.0f * c2 * c2 - 8.0f * c2 + 1.0f;  // cos(4t)
            const int   ki   = (co < r2) + (co < 0.0f) + (co < -r2) + (co <= -1.0f);
            const float sg   = (ki & 1) ? -1.0f : 1.0f;
            const float phi  = sg * cosm - 2.0f * (float)ki;
            float feat = co * xl;
            if (yy == myc) feat += (phi * xl - feat) * (1.0f / 1001.0f);
            __builtin_nontemporal_store(feat, out + (size_t)(m0 + rl) * Cc + myc);
        }
    }
}

// ---------------------------------------------------------------------------
extern "C" void kernel_launch(void* const* d_in, const int* in_sizes, int n_in,
                              void* d_out, int out_size, void* d_ws, size_t ws_size,
                              hipStream_t stream) {
    const float* x = (const float*)d_in[0];   // (512, 256)
    const float* w = (const float*)d_in[1];   // (100000, 256)
    const int*   y = (const int*)d_in[2];     // (512,)
    float* out = (float*)d_out;               // (512, 100000)

    float* inv_wl   = (float*)d_ws;           // C floats
    float* xlenv    = inv_wl + Cc;            // B floats
    float* inv_xlen = xlenv + Bb;             // B floats

    // 1) weight row inverse norms (also warms L2 with the weight matrix)
    wnorm_k<<<Cc / 8, 256, 0, stream>>>(w, inv_wl);
    // 2) x row norms
    xnorm_k<<<Bb / 8, 256, 0, stream>>>(x, xlenv, inv_xlen);
    // 3) fused GEMM + margin epilogue
    const size_t smem_bytes = 262144 + TM * (2 * sizeof(float) + sizeof(int)); // 265216 B
    dim3 grid(Cc / TN, Bb / TM);   // 3125 x 2
    asoft_main<<<grid, 256, smem_bytes, stream>>>(x, w, y, inv_wl, xlenv, inv_xlen, out);
}
